// MambaSSM_32315333935394
// MI455X (gfx1250) — matrix operand order repaired and checked by reading, not compiled, not run on hardware
//
#include <hip/hip_runtime.h>
#include <hip/hip_bf16.h>
#include <stdint.h>

#define D_MODEL 1024
#define D_STATE 16
#define D_CONV  4
#define D_INNER 2048
#define DT_RANK 64
#define BATCH   2
#define SEQ     2048
#define NROWS   (BATCH * SEQ)            // 4096
#define XDBL_N  (DT_RANK + 2 * D_STATE)  // 96

typedef __attribute__((ext_vector_type(16))) __bf16 bf16x16;
typedef __attribute__((ext_vector_type(8)))  __bf16 bf16x8;
typedef __attribute__((ext_vector_type(8)))  float  v8f;
typedef __attribute__((ext_vector_type(4)))  float  f32x4;
typedef __attribute__((ext_vector_type(4)))  int    i32x4;

struct bf16x8pair { bf16x8 lo, hi; };
__device__ __forceinline__ bf16x16 cat16(bf16x8 lo, bf16x8 hi) {
  bf16x8pair p{lo, hi};
  return __builtin_bit_cast(bf16x16, p);
}

// ---- CDNA5 async global->LDS copy (16B), with portable fallback ------------
__device__ __forceinline__ void async_copy16(__bf16* lds, const __bf16* gsrc) {
#if __has_builtin(__builtin_amdgcn_global_load_async_to_lds_b128)
  __builtin_amdgcn_global_load_async_to_lds_b128(
      (i32x4*)gsrc, (i32x4*)lds, 0, 0);
#else
  *(bf16x8*)lds = *(const bf16x8*)gsrc;
#endif
}

__device__ __forceinline__ void wait_async0() {
#if __has_builtin(__builtin_amdgcn_s_wait_asynccnt)
  __builtin_amdgcn_s_wait_asynccnt(0);
#elif __has_builtin(__builtin_amdgcn_global_load_async_to_lds_b128)
  asm volatile("s_wait_asynccnt 0x0" ::: "memory");
#endif
}

// ---------------------------------------------------------------------------
// Tiled bf16 WMMA GEMM:  out[M,N] = act(A[M,K] (lda, bf16) * W[N,K]^T (bf16))
// EPI 0: identity   EPI 1: softplus(v + bias[col])
// 128x128 block tile, BK=32, double-buffered LDS fed by async b128 copies.
// 8 waves, each computes 32x64 = 2x4 wmma tiles.
// ---------------------------------------------------------------------------
constexpr int BM = 128, BN = 128, BK = 32;
constexpr int LDSS = BK + 8;            // 80B row stride: conflict-free frags
constexpr int BUFE = BM * LDSS;         // elems per buffer

template<int EPI>
__global__ __launch_bounds__(256) void wmma_gemm_kernel(
    const __bf16* __restrict__ A, int lda,
    const __bf16* __restrict__ W,
    const float* __restrict__ bias,
    float* __restrict__ out,
    int M, int N, int K)
{
  __shared__ __align__(16) __bf16 As[2 * BUFE];
  __shared__ __align__(16) __bf16 Bs[2 * BUFE];

  const int tid   = threadIdx.x;
  const int lane  = tid & 31;
  const int wave  = tid >> 5;      // 0..7
  const int waveM = wave & 3;      // 4 waves along M
  const int waveN = wave >> 2;     // 2 waves along N
  const int rowTile = blockIdx.y * BM;
  const int colTile = blockIdx.x * BN;

  v8f zero = {};
  v8f acc[2][4];
#pragma unroll
  for (int i = 0; i < 2; ++i)
#pragma unroll
    for (int j = 0; j < 4; ++j) acc[i][j] = zero;

  // stage copier: 512 16B-chunks per (A,B) tile pair, 2 chunks/thread each
  auto issue_stage = [&](int k0, int buf) {
#pragma unroll
    for (int s = 0; s < 2; ++s) {
      const int c   = tid + s * 256;   // 0..511
      const int row = c >> 2;          // 0..127
      const int kc  = (c & 3) * 8;     // 0,8,16,24 (elems)
      async_copy16(As + buf * BUFE + row * LDSS + kc,
                   A + (size_t)(rowTile + row) * lda + k0 + kc);
      int n = colTile + row;
      if (n >= N) n = N - 1;           // clamp; dup cols discarded at store
      async_copy16(Bs + buf * BUFE + row * LDSS + kc,
                   W + (size_t)n * K + k0 + kc);
    }
  };

  const int nstages = K / BK;
  issue_stage(0, 0);

  for (int ks = 0; ks < nstages; ++ks) {
    const int buf = ks & 1;
    wait_async0();
    __syncthreads();                         // buf[ks&1] ready for everyone
    if (ks + 1 < nstages) issue_stage((ks + 1) * BK, buf ^ 1);

    // ---- fragments per ISA 7.12.2 layouts ----
    const int kbA = (lane >> 4) * 8;    // A: lanes>=16 hold K 8..15 / 24..31
    const int kbB = (lane >> 4) * 8;    // B: lanes>=16 hold K 8..15 / 24..31
    const __bf16* Ab = As + buf * BUFE;
    const __bf16* Bb = Bs + buf * BUFE;
    bf16x16 afrag[2], bfrag[4];
#pragma unroll
    for (int i = 0; i < 2; ++i) {
      const __bf16* p = Ab + (waveM * 32 + i * 16 + (lane & 15)) * LDSS;
      afrag[i] = cat16(*(const bf16x8*)(p + kbA), *(const bf16x8*)(p + kbA + 16));
    }
#pragma unroll
    for (int j = 0; j < 4; ++j) {
      const __bf16* p = Bb + (waveN * 64 + j * 16 + (lane & 15)) * LDSS + kbB;
      bfrag[j] = cat16(*(const bf16x8*)(p), *(const bf16x8*)(p + 16));
    }
#pragma unroll
    for (int i = 0; i < 2; ++i)
#pragma unroll
      for (int j = 0; j < 4; ++j)
        acc[i][j] = __builtin_amdgcn_wmma_f32_16x16x32_bf16(
            false, afrag[i], false, bfrag[j], (short)0, acc[i][j], false, false);
    __syncthreads();                         // done reading buf before rewrite
  }

  // ---- epilogue + store: C/D layout M = r + (lane>=16)*8, N = lane&15 ----
  const int mBase = rowTile + waveM * 32 + ((lane >> 4) << 3);
  const int nBase = colTile + waveN * 64 + (lane & 15);
#pragma unroll
  for (int i = 0; i < 2; ++i) {
#pragma unroll
    for (int j = 0; j < 4; ++j) {
      const int col = nBase + j * 16;
      if (col < N) {
        const float bv = (EPI == 1) ? bias[col] : 0.0f;
#pragma unroll
        for (int r = 0; r < 8; ++r) {
          const int row = mBase + i * 16 + r;
          float v = acc[i][j][r];
          if (EPI == 1) {
            v += bv;
            v = (v > 20.0f) ? v : __logf(1.0f + __expf(v));  // softplus
          }
          out[(size_t)row * N + col] = v;
        }
      }
    }
  }
}

// ---------------------------------------------------------------------------
// f32 -> bf16 bulk convert (8 elems / thread), native v_cvt via (__bf16) cast
// ---------------------------------------------------------------------------
__global__ __launch_bounds__(256) void cvt_bf16_kernel(
    const float* __restrict__ src, __bf16* __restrict__ dst, int n)
{
  int i = (blockIdx.x * 256 + threadIdx.x) * 8;
  if (i >= n) return;
  f32x4 a = *(const f32x4*)(src + i);
  f32x4 b = *(const f32x4*)(src + i + 4);
  bf16x8 o = { (__bf16)a.x, (__bf16)a.y, (__bf16)a.z, (__bf16)a.w,
               (__bf16)b.x, (__bf16)b.y, (__bf16)b.z, (__bf16)b.w };
  *(bf16x8*)(dst + i) = o;
}

// ---------------------------------------------------------------------------
// Causal depthwise conv1d (width 4, left-pad 3) + SiLU; writes f32 + bf16
// ---------------------------------------------------------------------------
__global__ __launch_bounds__(256) void conv_silu_kernel(
    const float* __restrict__ xz, const float* __restrict__ conv_w,
    const float* __restrict__ conv_b, float* __restrict__ xconv,
    __bf16* __restrict__ xconv_bf)
{
  int idx = blockIdx.x * 256 + threadIdx.x;          // (b*SEQ + l)*D_INNER + d
  int d  = idx & (D_INNER - 1);
  int bl = idx / D_INNER;
  int l  = bl & (SEQ - 1);
  float acc = conv_b[d];
#pragma unroll
  for (int j = 0; j < D_CONV; ++j) {
    int tl = l - (D_CONV - 1) + j;
    if (tl >= 0)
      acc = fmaf(conv_w[d * D_CONV + j],
                 xz[(size_t)(bl - l + tl) * (2 * D_INNER) + d], acc);
  }
  float s = acc / (1.0f + __expf(-acc));             // silu
  xconv[idx] = s;
  xconv_bf[idx] = (__bf16)s;
}

// ---------------------------------------------------------------------------
// Selective scan: one 16-lane half-wave per (b,d); lane n owns state n.
// ---------------------------------------------------------------------------
__global__ __launch_bounds__(256) void scan_kernel(
    const float* __restrict__ xdbl,   // [NROWS, 96]; B at 64..79, C at 80..95
    const float* __restrict__ dt,     // [NROWS, D_INNER] (post-softplus)
    const float* __restrict__ xconv,  // [NROWS, D_INNER]
    const float* __restrict__ A_log,  // [D_INNER, 16]
    const float* __restrict__ Dp,     // [D_INNER]
    float* __restrict__ y)            // [NROWS, D_INNER]
{
  const int gid = blockIdx.x * 256 + threadIdx.x;
  const int hw  = gid >> 4;                 // (b,d) pair, 0..4095
  const int n   = gid & 15;                 // state index
  const int b   = hw >> 11;                 // / D_INNER
  const int d   = hw & (D_INNER - 1);

  const float An = -__expf(A_log[d * D_STATE + n]);
  const float Dd = Dp[d];
  float state = 0.0f;

  for (int t = 0; t < SEQ; ++t) {
    const size_t row = (size_t)(b * SEQ + t);
    const float dtv = dt[row * D_INNER + d];
    const float xv  = xconv[row * D_INNER + d];
    const float Bv  = xdbl[row * XDBL_N + DT_RANK + n];
    const float Cv  = xdbl[row * XDBL_N + DT_RANK + D_STATE + n];
    const float dA  = __expf(dtv * An);
    state = fmaf(dA, state, dtv * Bv * xv);
    float p = state * Cv;
    p += __shfl_xor(p, 1, 32);
    p += __shfl_xor(p, 2, 32);
    p += __shfl_xor(p, 4, 32);
    p += __shfl_xor(p, 8, 32);
    if (n == 0) y[row * D_INNER + d] = fmaf(Dd, xv, p);
  }
}

// ---------------------------------------------------------------------------
// Gate: yg_bf = bf16( y * silu(z) ),  z = xz[:, 2048:4096]
// ---------------------------------------------------------------------------
__global__ __launch_bounds__(256) void gate_kernel(
    const float* __restrict__ xz, const float* __restrict__ y,
    __bf16* __restrict__ yg_bf)
{
  int idx = blockIdx.x * 256 + threadIdx.x;
  int d   = idx & (D_INNER - 1);
  int row = idx / D_INNER;
  float z = xz[(size_t)row * (2 * D_INNER) + D_INNER + d];
  yg_bf[idx] = (__bf16)(y[idx] * (z / (1.0f + __expf(-z))));
}

// ---------------------------------------------------------------------------
extern "C" void kernel_launch(void* const* d_in, const int* in_sizes, int n_in,
                              void* d_out, int out_size, void* d_ws, size_t ws_size,
                              hipStream_t stream) {
  const float* h        = (const float*)d_in[0];
  const float* in_w     = (const float*)d_in[1];
  const float* conv_w   = (const float*)d_in[2];
  const float* conv_b   = (const float*)d_in[3];
  const float* xproj_w  = (const float*)d_in[4];
  const float* dtproj_w = (const float*)d_in[5];
  const float* dtproj_b = (const float*)d_in[6];
  const float* A_log    = (const float*)d_in[7];
  const float* Dp       = (const float*)d_in[8];
  const float* out_w    = (const float*)d_in[9];
  float* out = (float*)d_out;

  // ---- workspace layout ----
  float* ws    = (float*)d_ws;
  float* xz    = ws;                                    // 4096 x 4096 f32
  float* xconv = xz    + (size_t)NROWS * 2 * D_INNER;   // 4096 x 2048 f32
  float* xdbl  = xconv + (size_t)NROWS * D_INNER;       // 4096 x 96   f32
  float* dtb   = xdbl  + (size_t)NROWS * XDBL_N;        // 4096 x 2048 f32
  float* ybuf  = dtb   + (size_t)NROWS * D_INNER;       // 4096 x 2048 f32
  __bf16* bfp     = (__bf16*)(ybuf + (size_t)NROWS * D_INNER);
  __bf16* h_bf    = bfp;                                  // 4096x1024
  __bf16* inw_bf  = h_bf    + (size_t)NROWS * D_MODEL;    // 4096x1024
  __bf16* xpw_bf  = inw_bf  + (size_t)2 * D_INNER * D_MODEL; // 96x2048
  __bf16* dtw_bf  = xpw_bf  + (size_t)XDBL_N * D_INNER;   // 2048x64
  __bf16* outw_bf = dtw_bf  + (size_t)D_INNER * DT_RANK;  // 1024x2048
  __bf16* xconv_bf= outw_bf + (size_t)D_MODEL * D_INNER;  // 4096x2048
  __bf16* xdbl_bf = xconv_bf+ (size_t)NROWS * D_INNER;    // 4096x96
  __bf16* yg_bf   = xdbl_bf + (size_t)NROWS * XDBL_N;     // 4096x2048

  dim3 blk(256);
  auto cvtBlocks = [](size_t n) { return (unsigned)(n / (256 * 8)); };

  // bf16 copies of GEMM operands (once per call, deterministic)
  cvt_bf16_kernel<<<cvtBlocks((size_t)NROWS * D_MODEL), blk, 0, stream>>>(h, h_bf, NROWS * D_MODEL);
  cvt_bf16_kernel<<<cvtBlocks((size_t)2 * D_INNER * D_MODEL), blk, 0, stream>>>(in_w, inw_bf, 2 * D_INNER * D_MODEL);
  cvt_bf16_kernel<<<cvtBlocks((size_t)XDBL_N * D_INNER), blk, 0, stream>>>(xproj_w, xpw_bf, XDBL_N * D_INNER);
  cvt_bf16_kernel<<<cvtBlocks((size_t)D_INNER * DT_RANK), blk, 0, stream>>>(dtproj_w, dtw_bf, D_INNER * DT_RANK);
  cvt_bf16_kernel<<<cvtBlocks((size_t)D_MODEL * D_INNER), blk, 0, stream>>>(out_w, outw_bf, D_MODEL * D_INNER);

  // xz = h @ in_proj_w^T        (4096 x 4096 x 1024)
  wmma_gemm_kernel<0><<<dim3(2 * D_INNER / BN, NROWS / BM), blk, 0, stream>>>(
      h_bf, D_MODEL, inw_bf, nullptr, xz, NROWS, 2 * D_INNER, D_MODEL);

  // x = silu(causal_conv(x))
  conv_silu_kernel<<<(NROWS * D_INNER) / 256, blk, 0, stream>>>(
      xz, conv_w, conv_b, xconv, xconv_bf);

  // x_dbl = x @ x_proj_w^T      (4096 x 96 x 2048)
  wmma_gemm_kernel<0><<<dim3(1, NROWS / BM), blk, 0, stream>>>(
      xconv_bf, D_INNER, xpw_bf, nullptr, xdbl, NROWS, XDBL_N, D_INNER);

  cvt_bf16_kernel<<<cvtBlocks((size_t)NROWS * XDBL_N), blk, 0, stream>>>(xdbl, xdbl_bf, NROWS * XDBL_N);

  // dt = softplus(x_dbl[:, :64] @ dt_proj_w^T + b)   (4096 x 2048 x 64)
  wmma_gemm_kernel<1><<<dim3(D_INNER / BN, NROWS / BM), blk, 0, stream>>>(
      xdbl_bf, XDBL_N, dtw_bf, dtproj_b, dtb, NROWS, D_INNER, DT_RANK);

  // selective scan
  scan_kernel<<<(BATCH * D_INNER * D_STATE) / 256, blk, 0, stream>>>(
      xdbl, dtb, xconv, A_log, Dp, ybuf);

  // yg = bf16(y * silu(z))
  gate_kernel<<<(NROWS * D_INNER) / 256, blk, 0, stream>>>(xz, ybuf, yg_bf);

  // out = yg @ out_proj_w^T     (4096 x 1024 x 2048)
  wmma_gemm_kernel<0><<<dim3(D_MODEL / BN, NROWS / BM), blk, 0, stream>>>(
      yg_bf, D_INNER, outw_bf, nullptr, out, NROWS, D_MODEL, D_INNER);
}
